// WeightGenerator_19430432047167
// MI455X (gfx1250) — matrix operand order repaired and checked by reading, not compiled, not run on hardware
//
#include <hip/hip_runtime.h>
#include <stdint.h>

// ---------- types for WMMA ----------
typedef __attribute__((ext_vector_type(16))) __bf16 v16bf;
typedef __attribute__((ext_vector_type(8)))  __bf16 v8bf;
typedef __attribute__((ext_vector_type(8)))  float  v8f;

__device__ __forceinline__ unsigned short f32_to_bf16(float f) {
  union { float f; uint32_t u; } c; c.f = f;
  uint32_t u = c.u;
  u += 0x7FFFu + ((u >> 16) & 1u);   // round-to-nearest-even
  return (unsigned short)(u >> 16);
}

// =====================================================================
// Weight swizzle: (Cout,64,3,3) f32 -> [NT][18][32][16] bf16 B-fragments
// GEMM K index = tap*64 + cin  (tap = ky*3+kx), split into 18 chunks of 32.
// B layout (16-bit, 32x16 KxN): lanes 0-15 hold kin {0-7,16-23} of col N=lane,
// lanes 16-31 hold kin {8-15,24-31} of col N=lane-16; 16 halves per lane.
// =====================================================================
__global__ __launch_bounds__(256) void prep_weights(
    const float* w0, const float* w1, const float* w2,
    const float* w3, const float* w4, const float* w5,
    unsigned short* o0, unsigned short* o1, unsigned short* o2,
    unsigned short* o3, unsigned short* o4, unsigned short* o5)
{
  const float* src; unsigned short* dst; int Cout, NT;
  switch (blockIdx.y) {
    case 0:  src = w0; dst = o0; Cout = 64; NT = 4; break;
    case 1:  src = w1; dst = o1; Cout = 64; NT = 4; break;
    case 2:  src = w2; dst = o2; Cout = 1;  NT = 1; break;
    case 3:  src = w3; dst = o3; Cout = 64; NT = 4; break;
    case 4:  src = w4; dst = o4; Cout = 64; NT = 4; break;
    default: src = w5; dst = o5; Cout = 1;  NT = 1; break;
  }
  const int total = NT * 18 * 32 * 16;
  for (int e = blockIdx.x * 256 + threadIdx.x; e < total; e += gridDim.x * 256) {
    int j    = e & 15;
    int lane = (e >> 4) & 31;
    int kc   = (e >> 9) % 18;
    int nt   = e / (512 * 18);
    int n    = nt * 16 + (lane & 15);
    int kin  = ((j >> 3) << 4) + ((lane >> 4) << 3) + (j & 7);
    int K    = kc * 32 + kin;          // 0..575
    int cin  = K & 63;
    int tap  = K >> 6;                 // 0..8
    float v  = (n < Cout) ? src[((n * 64) + cin) * 9 + tap] : 0.f;
    dst[e] = f32_to_bf16(v);
  }
}

// =====================================================================
// 3x3 SAME conv, Cin=64, via implicit GEMM on v_wmma_f32_16x16x32_bf16.
// Block = 256 thr (8 waves). blockIdx = (x-strip of 128, row y, image z).
// Each wave: M=16 pixels, N=16*NT, K=576 (18 chunks of 32).
// NT/COUT are compile-time so the inner loop is branch-free (no acc moves).
// =====================================================================
template <int NT, int COUT>
__global__ __launch_bounds__(256) void conv3x3_wmma(
    const float* __restrict__ in,  long inImgStride,
    float* __restrict__ out,       long outImgStride,
    const unsigned short* __restrict__ wsw,   // [NT][18][32][16] bf16
    const float* __restrict__ bias,
    int H, int W)
{
  __shared__ alignas(16) unsigned short lds[3 * 130 * 64];   // 49,920 B
  const int tid  = threadIdx.x;
  const int x0b  = blockIdx.x * 128;
  const int y    = blockIdx.y;
  const long zin  = (long)blockIdx.z * inImgStride;
  const long zout = (long)blockIdx.z * outImgStride;
  const int  HW   = H * W;

  // hint: weight fragments are hot & shared by every block
  __builtin_prefetch(wsw, 0, 1);

  // ---- stage 3-row halo strip (130 x, 64 ch) as bf16, cin-contiguous ----
  for (int t = tid; t < 3 * 130 * 64; t += 256) {
    int ky  = t / (64 * 130);
    int rem = t - ky * (64 * 130);
    int cin = rem / 130;
    int xl  = rem - cin * 130;          // x fast -> coalesced global reads
    int xs  = x0b + xl - 1;
    int ys  = y + ky - 1;
    float v = 0.f;
    if (xs >= 0 && xs < W && ys >= 0 && ys < H)
      v = in[zin + (long)cin * HW + (long)ys * W + xs];
    lds[(ky * 130 + xl) * 64 + cin] = f32_to_bf16(v);
  }
  __syncthreads();

  const int wv   = tid >> 5;          // wave -> 16-pixel tile
  const int lane = tid & 31;
  const int m    = lane & 15;
  const int kh   = lane >> 4;         // A-layout K-half select

  v8f acc[NT] = {};

#pragma unroll
  for (int kc = 0; kc < 18; ++kc) {
    const int tap  = kc >> 1;
    const int cin0 = (kc & 1) << 5;
    const int ky   = tap / 3;
    const int kx   = tap - ky * 3;
    const int xl   = wv * 16 + m + kx;                 // (x+kx-1)-x0b+1
    const int base = (ky * 130 + xl) * 64 + cin0;
    // 16-bit A 16x32 layout: lane<16 -> kin {0-7,16-23}; lane>=16 -> {8-15,24-31}
    v8bf alo = *reinterpret_cast<const v8bf*>(&lds[base + kh * 8]);
    v8bf ahi = *reinterpret_cast<const v8bf*>(&lds[base + 16 + kh * 8]);
    v16bf a = __builtin_shufflevector(alo, ahi,
              0, 1, 2, 3, 4, 5, 6, 7, 8, 9, 10, 11, 12, 13, 14, 15);
#pragma unroll
    for (int nt = 0; nt < NT; ++nt) {
      v16bf b = *reinterpret_cast<const v16bf*>(
          &wsw[(((size_t)nt * 18 + kc) * 32 + lane) * 16]);
      acc[nt] = __builtin_amdgcn_wmma_f32_16x16x32_bf16(
          false, a, false, b, (short)0, acc[nt], false, false);
    }
  }

  // ---- epilogue: C/D layout: lane -> N=lane%16, M(r)=r+8*(lane/16) ----
  const int n  = lane & 15;
  const int mb = (lane >> 4) * 8;
#pragma unroll
  for (int nt = 0; nt < NT; ++nt) {
    int cout = nt * 16 + n;
    if (cout < COUT) {
      float bv = bias[cout];
#pragma unroll
      for (int r = 0; r < 8; ++r) {
        int x = x0b + wv * 16 + mb + r;
        float v = acc[nt][r] + bv;
        v = v > 0.f ? v : 0.f;
        out[zout + (long)cout * HW + (long)y * W + x] = v;
      }
    }
  }
}

// =====================================================================
// Fused resize + pad + softmax + feature blend.
// =====================================================================
__device__ __forceinline__ float bilin128(const float* __restrict__ s,
                                          int x0, int y0, float fx, float fy) {
  const float* r0 = s + y0 * 128 + x0;
  float a = r0[0], b = r0[1], c = r0[128], d = r0[129];
  float top = a + (b - a) * fx;
  float bot = c + (d - c) * fx;
  return top + (bot - top) * fy;
}

__global__ __launch_bounds__(256) void fuse_kernel(
    const float* __restrict__ inst,    // (8,64,128,128)
    const float* __restrict__ bg,      // (64,512,512)
    const float* __restrict__ masks,   // (8,128,128)
    const float* __restrict__ bgmask,  // (512,512)
    float* __restrict__ out)           // (64,512,512)
{
  const int p = blockIdx.x * 256 + threadIdx.x;   // 0..262143
  const int x = p & 511, y = p >> 9;
  // BOX_INFO columns: l, r, t, btm, bw, bh
  constexpr int BL[8]  = {0, 64, 128, 300, 64, 200, 32, 256};
  constexpr int BT[8]  = {0, 32, 256, 300, 128, 100, 256, 320};
  constexpr int BWD[8] = {256, 192, 320, 128, 384, 256, 160, 224};
  constexpr int BHT[8] = {256, 320, 192, 128, 256, 384, 224, 160};

  bool  inside[8];
  int   ix0[8], iy0[8];
  float fx[8], fy[8];
  float logit[9];
  bool  any = false;

#pragma unroll
  for (int i = 0; i < 8; ++i) {
    int lx = x - BL[i], ly = y - BT[i];
    bool in = (lx >= 0) && (lx < BWD[i]) && (ly >= 0) && (ly < BHT[i]);
    inside[i] = in;
    any = any || in;
    float li = 0.f;
    int xi = 0, yi = 0; float fu = 0.f, fv = 0.f;
    if (in) {
      float u = (lx + 0.5f) * (128.f / BWD[i]) - 0.5f;
      float v = (ly + 0.5f) * (128.f / BHT[i]) - 0.5f;
      xi = (int)floorf(u); fu = u - xi;
      yi = (int)floorf(v); fv = v - yi;
      if (xi < 0)   { xi = 0;   fu = 0.f; }
      if (xi > 126) { xi = 126; fu = 1.f; }
      if (yi < 0)   { yi = 0;   fv = 0.f; }
      if (yi > 126) { yi = 126; fv = 1.f; }
      li = bilin128(masks + (size_t)i * 16384, xi, yi, fu, fv);
    }
    ix0[i] = xi; iy0[i] = yi; fx[i] = fu; fy[i] = fv;
    logit[i] = li;               // zero-padded outside box (matches jnp.pad)
  }
  logit[8] = bgmask[p] + (any ? 0.f : 100000.f);

  float mx = logit[0];
#pragma unroll
  for (int i = 1; i < 9; ++i) mx = fmaxf(mx, logit[i]);
  float prob[9]; float s = 0.f;
#pragma unroll
  for (int i = 0; i < 9; ++i) { prob[i] = __expf(logit[i] - mx); s += prob[i]; }
  float invs = 1.f / s;
#pragma unroll
  for (int i = 0; i < 9; ++i) prob[i] *= invs;

  for (int c = 0; c < 64; ++c) {
    float acc = prob[8] * bg[(size_t)c * 262144 + p];
#pragma unroll
    for (int i = 0; i < 8; ++i) {
      if (inside[i]) {
        const float* src = inst + ((size_t)i * 64 + c) * 16384;
        acc += prob[i] * bilin128(src, ix0[i], iy0[i], fx[i], fy[i]);
      }
    }
    out[(size_t)c * 262144 + p] = acc;
  }
}

// =====================================================================
// Host launcher
// =====================================================================
extern "C" void kernel_launch(void* const* d_in, const int* in_sizes, int n_in,
                              void* d_out, int out_size, void* d_ws, size_t ws_size,
                              hipStream_t stream) {
  (void)in_sizes; (void)n_in; (void)out_size; (void)ws_size;
  const float* inst = (const float*)d_in[0];
  const float* bg   = (const float*)d_in[1];
  const float* iw1  = (const float*)d_in[2];  const float* ib1 = (const float*)d_in[3];
  const float* iw2  = (const float*)d_in[4];  const float* ib2 = (const float*)d_in[5];
  const float* iw3  = (const float*)d_in[6];  const float* ib3 = (const float*)d_in[7];
  const float* bw1  = (const float*)d_in[8];  const float* bb1 = (const float*)d_in[9];
  const float* bw2  = (const float*)d_in[10]; const float* bb2 = (const float*)d_in[11];
  const float* bw3  = (const float*)d_in[12]; const float* bb3 = (const float*)d_in[13];
  float* out = (float*)d_out;

  char* ws = (char*)d_ws;
  size_t off = 0;
  auto alloc = [&](size_t bytes) -> void* {
    void* p = ws + off;
    off = (off + bytes + 255) & ~(size_t)255;
    return p;
  };
  const size_t W64SZ = (size_t)4 * 18 * 32 * 16 * 2;   // 73,728 B
  const size_t W1SZ  = (size_t)1 * 18 * 32 * 16 * 2;   // 18,432 B
  unsigned short* swi1 = (unsigned short*)alloc(W64SZ);
  unsigned short* swi2 = (unsigned short*)alloc(W64SZ);
  unsigned short* swi3 = (unsigned short*)alloc(W1SZ);
  unsigned short* swb1 = (unsigned short*)alloc(W64SZ);
  unsigned short* swb2 = (unsigned short*)alloc(W64SZ);
  unsigned short* swb3 = (unsigned short*)alloc(W1SZ);
  float* t1    = (float*)alloc((size_t)64 * 512 * 512 * 4);   // 64 MB
  float* t2    = (float*)alloc((size_t)64 * 512 * 512 * 4);   // 64 MB
  float* masks = (float*)alloc((size_t)8 * 128 * 128 * 4);
  float* bgm   = (float*)alloc((size_t)512 * 512 * 4);

  // 1) weight swizzle (6 layers)
  prep_weights<<<dim3(144, 6), 256, 0, stream>>>(
      iw1, iw2, iw3, bw1, bw2, bw3, swi1, swi2, swi3, swb1, swb2, swb3);

  const long ISTR = (long)64 * 128 * 128;   // per-instance 64ch stride
  // 2) instance conv block (8 images batched on grid.z)
  conv3x3_wmma<4, 64><<<dim3(1, 128, 8), 256, 0, stream>>>(
      inst, ISTR, t1, ISTR, swi1, ib1, 128, 128);
  conv3x3_wmma<4, 64><<<dim3(1, 128, 8), 256, 0, stream>>>(
      t1, ISTR, t2, ISTR, swi2, ib2, 128, 128);
  conv3x3_wmma<1, 1><<<dim3(1, 128, 8), 256, 0, stream>>>(
      t2, ISTR, masks, (long)128 * 128, swi3, ib3, 128, 128);

  // 3) background conv block (512x512)
  conv3x3_wmma<4, 64><<<dim3(4, 512, 1), 256, 0, stream>>>(
      bg, 0, t1, 0, swb1, bb1, 512, 512);
  conv3x3_wmma<4, 64><<<dim3(4, 512, 1), 256, 0, stream>>>(
      t1, 0, t2, 0, swb2, bb2, 512, 512);
  conv3x3_wmma<1, 1><<<dim3(4, 512, 1), 256, 0, stream>>>(
      t2, 0, bgm, 0, swb3, bb3, 512, 512);

  // 4) fused resize/pad/softmax/blend
  fuse_kernel<<<dim3(1024), 256, 0, stream>>>(inst, bg, masks, bgm, out);
}
